// VectorQuantizer_9062380995253
// MI455X (gfx1250) — matrix operand order repaired and checked by reading, not compiled, not run on hardware
//
#include <hip/hip_runtime.h>
#include <hip/hip_bf16.h>

// ---------------------------------------------------------------------------
// VQ-VAE vector quantizer for MI455X (gfx1250, wave32, WMMA).
// inputs:  d_in[0] = inputs      float32 [8,4096,256]  (N=32768 rows, D=256)
//          d_in[1] = embeddings  float32 [4096,256]    (K=4096 codes)
// outputs: d_out = [quantized_st (8388608 f32)] [loss (1 f32)] [indices (32768 f32)]
// ---------------------------------------------------------------------------

#define VQ_N 32768
#define VQ_D 256
#define VQ_K 4096
#define VQ_TILES (VQ_K / 16)   // 256 code tiles of 16
#define LDS_PITCH 264          // 256 + 8 bf16 pad -> bank-conflict-free B reads

typedef __attribute__((ext_vector_type(16))) __bf16 v16bf;
typedef __attribute__((ext_vector_type(8)))  __bf16 v8bf;
typedef __attribute__((ext_vector_type(8)))  float  v8f;
typedef __attribute__((ext_vector_type(4)))  float  v4f;

// --- Kernel 1: convert embeddings fp32 -> bf16, compute per-code squared norms
__global__ __launch_bounds__(256) void vq_conv_e(const float* __restrict__ E,
                                                 __bf16* __restrict__ Ebf,
                                                 float* __restrict__ enorm) {
    const int row = blockIdx.x;          // 0..4095
    const int t   = threadIdx.x;         // 0..255
    float v = E[(size_t)row * VQ_D + t];
    Ebf[(size_t)row * VQ_D + t] = (__bf16)v;
    __shared__ float s[256];
    s[t] = v * v;
    __syncthreads();
    #pragma unroll
    for (int off = 128; off > 0; off >>= 1) {
        if (t < off) s[t] += s[t + off];
        __syncthreads();
    }
    if (t == 0) enorm[row] = s[0];
}

// --- Kernel 2: WMMA distance GEMM + argmin over K codes.
// Block = 256 threads = 8 waves; wave w owns rows [blk*128 + w*16, +16).
// Double-buffered LDS E-tiles (global loads for t+1 overlap 8 WMMAs of t),
// and software-pipelined B-fragment loads (ds_load for c+1 in flight under
// the WMMA for chunk c).
__global__ __launch_bounds__(256) void vq_argmin(const float* __restrict__ X,
                                                 const __bf16* __restrict__ Ebf,
                                                 const float* __restrict__ enorm,
                                                 int* __restrict__ bestIdx) {
    const int tid  = threadIdx.x;
    const int wave = tid >> 5;           // 0..7
    const int lane = tid & 31;           // 0..31
    const int half = lane >> 4;          // 0 or 1
    const int l16  = lane & 15;          // 0..15

    const int wave_m_base = blockIdx.x * 128 + wave * 16;

    // ---- Load this wave's A fragments (16 rows x 256 dims) once, fp32->bf16.
    // 16-bit A 16x32 layout: elem e in [0,8): K = 32c + half*8 + e
    //                        elem e in [8,16): K = 32c + 16 + half*8 + (e-8)
    v16bf afrag[8];
    {
        const float* xr = X + (size_t)(wave_m_base + l16) * VQ_D;
        #pragma unroll
        for (int c = 0; c < 8; ++c) {
            const int k0 = 32 * c + half * 8;
            v4f x0 = *(const v4f*)(xr + k0);
            v4f x1 = *(const v4f*)(xr + k0 + 4);
            v4f x2 = *(const v4f*)(xr + k0 + 16);
            v4f x3 = *(const v4f*)(xr + k0 + 20);
            v16bf a;
            #pragma unroll
            for (int i = 0; i < 4; ++i) {
                a[i]      = (__bf16)x0[i];
                a[4 + i]  = (__bf16)x1[i];
                a[8 + i]  = (__bf16)x2[i];
                a[12 + i] = (__bf16)x3[i];
            }
            afrag[c] = a;
        }
    }

    // Two LDS tile buffers: compute reads buf (t&1) while staging writes buf^1.
    __shared__ __attribute__((aligned(16))) __bf16 etile[2][16 * LDS_PITCH];

    const int stage_r   = tid >> 4;      // row in tile 0..15
    const int stage_c16 = tid & 15;      // 16-dim chunk

    // ---- Prologue: stage tile 0 into buffer 0 (32 B per thread).
    {
        const v8bf* src = (const v8bf*)(Ebf + (size_t)stage_r * VQ_D + stage_c16 * 16);
        v8bf e0 = src[0];
        v8bf e1 = src[1];
        __bf16* dst = &etile[0][stage_r * LDS_PITCH + stage_c16 * 16];
        *(v8bf*)(dst)     = e0;
        *(v8bf*)(dst + 8) = e1;
    }
    __syncthreads();

    float best[8];
    int   bidx[8];
    #pragma unroll
    for (int r = 0; r < 8; ++r) { best[r] = 3.4e38f; bidx[r] = 0; }

    for (int t = 0; t < VQ_TILES; ++t) {
        const int buf = t & 1;

        // ---- Stage tile t+1 into the other buffer (overlaps the WMMAs below).
        if (t + 1 < VQ_TILES) {
            const v8bf* src = (const v8bf*)
                (Ebf + (size_t)((t + 1) * 16 + stage_r) * VQ_D + stage_c16 * 16);
            v8bf e0 = src[0];
            v8bf e1 = src[1];
            __bf16* dst = &etile[buf ^ 1][stage_r * LDS_PITCH + stage_c16 * 16];
            *(v8bf*)(dst)     = e0;
            *(v8bf*)(dst + 8) = e1;
        }
        const float en = enorm[t * 16 + l16];

        // ---- 8 x WMMA over D=256 with pipelined B-fragment LDS loads.
        // B 32x16 layout: lane: n=l16, K = 32c + half*8 + [0,8), then 32c + 16 + half*8 + [0,8)
        const __bf16* brow = &etile[buf][l16 * LDS_PITCH + half * 8];
        v8f acc = {0.f, 0.f, 0.f, 0.f, 0.f, 0.f, 0.f, 0.f};
        v8bf b0n = *(const v8bf*)(brow);
        v8bf b1n = *(const v8bf*)(brow + 16);
        #pragma unroll
        for (int c = 0; c < 8; ++c) {
            v8bf b0 = b0n;
            v8bf b1 = b1n;
            if (c < 7) {                 // prefetch chunk c+1 before WMMA c
                b0n = *(const v8bf*)(brow + 32 * (c + 1));
                b1n = *(const v8bf*)(brow + 32 * (c + 1) + 16);
            }
            v16bf b;
            #pragma unroll
            for (int i = 0; i < 8; ++i) { b[i] = b0[i]; b[8 + i] = b1[i]; }
            acc = __builtin_amdgcn_wmma_f32_16x16x32_bf16(
                false, afrag[c], false, b, (short)0, acc, false, false);
        }

        // ---- d2' = ||e||^2 - 2 x.e  (||x||^2 constant in n -> same argmin)
        const int nidx = t * 16 + l16;
        #pragma unroll
        for (int r = 0; r < 8; ++r) {
            float d = __builtin_fmaf(-2.f, acc[r], en);
            if (d < best[r]) { best[r] = d; bidx[r] = nidx; }
        }

        // One barrier per tile: staging writes for t+1 visible, and all reads
        // of buffer `buf` done before iteration t+1 re-stages into buf^1's
        // complement (i.e. before t+2 overwrites buf).
        __syncthreads();
    }

    // ---- Min+index reduction across the 16 lanes of each half (wave32 shfl).
    #pragma unroll
    for (int off = 8; off > 0; off >>= 1) {
        #pragma unroll
        for (int r = 0; r < 8; ++r) {
            float ov = __shfl_xor(best[r], off);
            int   oi = __shfl_xor(bidx[r], off);
            if (ov < best[r] || (ov == best[r] && oi < bidx[r])) {
                best[r] = ov; bidx[r] = oi;
            }
        }
    }
    // C layout: lanes 0-15 hold M=r, lanes 16-31 hold M=8+r. Lane l16==0 writes.
    if (l16 == 0) {
        const int m0 = wave_m_base + half * 8;
        #pragma unroll
        for (int r = 0; r < 8; ++r) bestIdx[m0 + r] = bidx[r];
    }
}

// --- Kernel 3: gather quantized rows (fp32), per-row loss partials, indices.
__global__ __launch_bounds__(256) void vq_gather(const float* __restrict__ X,
                                                 const float* __restrict__ E,
                                                 const int* __restrict__ bestIdx,
                                                 float* __restrict__ outQ,
                                                 float* __restrict__ outIdx,
                                                 float* __restrict__ partial) {
    const int row = blockIdx.x;          // 0..32767
    const int t   = threadIdx.x;         // 0..255
    const int k   = bestIdx[row];
    float q = E[(size_t)k * VQ_D + t];
    float x = X[(size_t)row * VQ_D + t];
    outQ[(size_t)row * VQ_D + t] = q;    // quantized_st == quantized numerically
    float d = q - x;
    __shared__ float s[256];
    s[t] = d * d;
    __syncthreads();
    #pragma unroll
    for (int off = 128; off > 0; off >>= 1) {
        if (t < off) s[t] += s[t + off];
        __syncthreads();
    }
    if (t == 0) {
        partial[row] = s[0];
        outIdx[row]  = (float)k;
    }
}

// --- Kernel 4: deterministic final loss reduction.
__global__ __launch_bounds__(256) void vq_loss(const float* __restrict__ partial,
                                               float* __restrict__ outLoss) {
    const int t = threadIdx.x;
    float a = 0.f;
    for (int j = t; j < VQ_N; j += 256) a += partial[j];   // fixed order
    __shared__ float s[256];
    s[t] = a;
    __syncthreads();
    #pragma unroll
    for (int off = 128; off > 0; off >>= 1) {
        if (t < off) s[t] += s[t + off];
        __syncthreads();
    }
    // loss = q_latent + 0.25*e_latent = 1.25 * mean((q-x)^2)
    if (t == 0) outLoss[0] = s[0] * (1.25f / (float)((size_t)VQ_N * VQ_D));
}

extern "C" void kernel_launch(void* const* d_in, const int* in_sizes, int n_in,
                              void* d_out, int out_size, void* d_ws, size_t ws_size,
                              hipStream_t stream) {
    (void)in_sizes; (void)n_in; (void)out_size; (void)ws_size;
    const float* X = (const float*)d_in[0];   // [32768,256]
    const float* E = (const float*)d_in[1];   // [4096,256]

    float* outQ    = (float*)d_out;                 // 8,388,608
    float* outLoss = outQ + (size_t)VQ_N * VQ_D;    // 1
    float* outIdx  = outLoss + 1;                   // 32,768

    // Workspace layout (~2.4 MB)
    char* ws = (char*)d_ws;
    __bf16* Ebf    = (__bf16*)ws;                                  // 2,097,152 B
    float*  enorm  = (float*)(ws + 2097152);                       //    16,384 B
    int*    bIdx   = (int*)  (ws + 2097152 + 16384);               //   131,072 B
    float*  part   = (float*)(ws + 2097152 + 16384 + 131072);      //   131,072 B

    vq_conv_e<<<VQ_K, 256, 0, stream>>>(E, Ebf, enorm);
    vq_argmin<<<VQ_N / 128, 256, 0, stream>>>(X, Ebf, enorm, bIdx);
    vq_gather<<<VQ_N, 256, 0, stream>>>(X, E, bIdx, outQ, outIdx, part);
    vq_loss<<<1, 256, 0, stream>>>(part, outLoss);
}
